// learnable_sig_59459527246613
// MI455X (gfx1250) — matrix operand order repaired and checked by reading, not compiled, not run on hardware
//
#include <hip/hip_runtime.h>
#include <hip/hip_bf16.h>

typedef __attribute__((ext_vector_type(16))) __bf16 v16bf;
typedef __attribute__((ext_vector_type(8)))  float  v8f;

// ---------------------------------------------------------------------------
// Kernel 1: init per-graph output table gout[G][4] = (b1_2[0], b1_2[1],
// b2_2[0], b2_2[1]).  GEMM kernel accumulates on top of the biases.
// ---------------------------------------------------------------------------
__global__ void init_gout_kernel(const float* __restrict__ b12,
                                 const float* __restrict__ b22,
                                 float* __restrict__ gout, int G) {
    int g = blockIdx.x * blockDim.x + threadIdx.x;
    if (g < G) {
        ((float4*)gout)[g] = make_float4(b12[0], b12[1], b22[0], b22[1]);
    }
}

// ---------------------------------------------------------------------------
// Kernel 2: fused per-graph 2-layer MLP head.
//   H = relu(xcore @ W + b)            via bf16x3 WMMA (f32 accumulate)
//   gout[g][which*2 + c] += sum_n H[g][n] * Wc[n][c]
// One wave (32 threads) computes one 16x16 output tile of H, contracts it
// against the 2-column second layer in registers, lane-reduces, atomically
// accumulates.  grid = (F/16, G/16, 2), block = 32.
// ---------------------------------------------------------------------------
__device__ __forceinline__ void bf16_split(float v, __bf16& hi, __bf16& lo) {
    hi = (__bf16)v;
    lo = (__bf16)(v - (float)hi);
}

__global__ void gemm_fused_kernel(const float* __restrict__ xcore,
                                  const float* __restrict__ W1,
                                  const float* __restrict__ b1,
                                  const float* __restrict__ Wc1,
                                  const float* __restrict__ W2,
                                  const float* __restrict__ b2,
                                  const float* __restrict__ Wc2,
                                  float* __restrict__ gout, int F) {
    const int which = blockIdx.z;
    const float* __restrict__ W  = which ? W2  : W1;
    const float* __restrict__ bb = which ? b2  : b1;
    const float* __restrict__ Wc = which ? Wc2 : Wc1;

    const int tile_n = blockIdx.x;            // feature tile  (F/16)
    const int tile_m = blockIdx.y;            // graph tile    (G/16)
    const int lane   = threadIdx.x;           // 0..31 (wave32)
    const int half   = lane >> 4;             // 0 or 1
    const int lq     = lane & 15;

    const int arow = tile_m * 16 + lq;        // graph row (A: M = lq)
    const int bcol = tile_n * 16 + lq;        // feature col (B: N = lq)

    v8f acc = {};
    const int kblks = F >> 5;                 // K tiles of 32

    for (int kb = 0; kb < kblks; ++kb) {
        const int k0 = kb << 5;

        // ---- A fragment: 16x32 bf16. lane half selects K groups:
        //   j in [0,8)  -> K = k0 + half*8 + j
        //   j in [8,16) -> K = k0 + 16 + half*8 + (j-8)
        const float* arp = xcore + (size_t)arow * F + k0 + half * 8;
        float4 a0 = ((const float4*)arp)[0];
        float4 a1 = ((const float4*)arp)[1];
        float4 a2 = ((const float4*)(arp + 16))[0];
        float4 a3 = ((const float4*)(arp + 16))[1];
        float av[16] = { a0.x, a0.y, a0.z, a0.w, a1.x, a1.y, a1.z, a1.w,
                         a2.x, a2.y, a2.z, a2.w, a3.x, a3.y, a3.z, a3.w };

        // ---- B fragment: 32x16 bf16. same K groups as A, N = bcol
        float bv[16];
        const float* wp = W + (size_t)(k0 + half * 8) * F + bcol;
        #pragma unroll
        for (int j = 0; j < 16; ++j) bv[j] = wp[(size_t)(j < 8 ? j : j + 8) * F];

        // ---- split to bf16 hi/lo (bf16x3 compensated product)
        v16bf aHi, aLo, bHi, bLo;
        #pragma unroll
        for (int j = 0; j < 16; ++j) {
            __bf16 h, l;
            bf16_split(av[j], h, l); aHi[j] = h; aLo[j] = l;
            bf16_split(bv[j], h, l); bHi[j] = h; bLo[j] = l;
        }

        acc = __builtin_amdgcn_wmma_f32_16x16x32_bf16(
                  false, aHi, false, bHi, (short)0, acc, false, false);
        acc = __builtin_amdgcn_wmma_f32_16x16x32_bf16(
                  false, aHi, false, bLo, (short)0, acc, false, false);
        acc = __builtin_amdgcn_wmma_f32_16x16x32_bf16(
                  false, aLo, false, bHi, (short)0, acc, false, false);
    }

    // Epilogue: bias + relu, contract against the 2 second-layer columns,
    // reduce across the 16 lanes sharing a row half, accumulate into gout.
    const float bias = bb[bcol];
    const float w2c0 = Wc[bcol * 2 + 0];
    const float w2c1 = Wc[bcol * 2 + 1];

    #pragma unroll
    for (int r = 0; r < 8; ++r) {
        float h = acc[r] + bias;
        h = h > 0.0f ? h : 0.0f;
        float v0 = h * w2c0;
        float v1 = h * w2c1;
        #pragma unroll
        for (int m = 8; m >= 1; m >>= 1) {
            v0 += __shfl_xor(v0, m, 32);
            v1 += __shfl_xor(v1, m, 32);
        }
        if (lq == 0) {
            const int g = tile_m * 16 + half * 8 + r;   // row M = r + 8*half
            atomicAdd(&gout[g * 4 + which * 2 + 0], v0);
            atomicAdd(&gout[g * 4 + which * 2 + 1], v1);
        }
    }
}

// ---------------------------------------------------------------------------
// Kernel 3: bandwidth-bound node stage.  2 nodes per thread, 128-bit ld/st.
//   out[i] = sigmoid(x[i] * (2 + h2[batch[i]]) - h1[batch[i]])
// ---------------------------------------------------------------------------
__device__ __forceinline__ float sigmoidf(float t) {
    return 1.0f / (1.0f + __expf(-t));
}

__global__ void node_kernel(const float* __restrict__ x,
                            const int* __restrict__ batch,
                            const float* __restrict__ gout,
                            float* __restrict__ out, int npair) {
    int i = blockIdx.x * blockDim.x + threadIdx.x;
    if (i >= npair) return;
    float4 xv = ((const float4*)x)[i];        // nodes 2i and 2i+1
    int2  bi  = ((const int2*)batch)[i];
    float4 p0 = ((const float4*)gout)[bi.x];  // (h1_0, h1_1, h2_0, h2_1)
    float4 p1 = ((const float4*)gout)[bi.y];
    float4 o;
    o.x = sigmoidf(xv.x * (2.0f + p0.z) - p0.x);
    o.y = sigmoidf(xv.y * (2.0f + p0.w) - p0.y);
    o.z = sigmoidf(xv.z * (2.0f + p1.z) - p1.x);
    o.w = sigmoidf(xv.w * (2.0f + p1.w) - p1.y);
    ((float4*)out)[i] = o;
}

// ---------------------------------------------------------------------------
extern "C" void kernel_launch(void* const* d_in, const int* in_sizes, int n_in,
                              void* d_out, int out_size, void* d_ws, size_t ws_size,
                              hipStream_t stream) {
    const float* x     = (const float*)d_in[0];
    const float* xcore = (const float*)d_in[1];
    const int*   batch = (const int*)  d_in[2];
    const float* W1_1  = (const float*)d_in[3];
    const float* b1_1  = (const float*)d_in[4];
    const float* W1_2  = (const float*)d_in[5];
    const float* b1_2  = (const float*)d_in[6];
    const float* W2_1  = (const float*)d_in[7];
    const float* b2_1  = (const float*)d_in[8];
    const float* W2_2  = (const float*)d_in[9];
    const float* b2_2  = (const float*)d_in[10];

    const int F = in_sizes[4];            // 1024
    const int G = in_sizes[1] / F;        // 512
    const int N = in_sizes[2];            // 16777216

    float* gout = (float*)d_ws;           // G x 4 floats (8 KB)

    init_gout_kernel<<<(G + 255) / 256, 256, 0, stream>>>(b1_2, b2_2, gout, G);

    dim3 grid(F / 16, G / 16, 2);
    gemm_fused_kernel<<<grid, 32, 0, stream>>>(xcore,
                                               W1_1, b1_1, W1_2,
                                               W2_1, b2_1, W2_2,
                                               gout, F);

    const int npair = N / 2;
    node_kernel<<<(npair + 255) / 256, 256, 0, stream>>>(x, batch, gout,
                                                         (float*)d_out, npair);
}
